// Classifier_35424890257572
// MI455X (gfx1250) — matrix operand order repaired and checked by reading, not compiled, not run on hardware
//
#include <hip/hip_runtime.h>

// GAT classifier for MI455X (gfx1250, wave32).
// GEMMs run on v_wmma_f32_16x16x32_f16 (f16 in / fp32 accum); edge softmax +
// scatter-aggregate use fp32 L2 atomics (the HBM-bound part: ~6 GB of
// gather+scatter traffic -> ~0.3 ms at 23.3 TB/s; 12 GFLOP of GEMM is noise
// once it's on the matrix pipes).
// Workspace use: ~124 MB (buffers are aliased across the two GAT layers).

typedef __attribute__((ext_vector_type(16))) _Float16 v16h;
typedef __attribute__((ext_vector_type(8)))  float    v8f;

union ABFrag { v16h v; float4 f[2]; };

// ---------------- utility kernels ----------------

__global__ void fill_f32(float* __restrict__ p, float v, int n) {
  int i = blockIdx.x * blockDim.x + threadIdx.x;
  if (i < n) p[i] = v;
}

// convert fp32 [srcRows, srcCols] (+optional bias per column) to zero-padded
// f16 [dstRows, dstLd]; fully initializes dst (pad rows/cols are 0).
__global__ void cvt_pad_f16(const float* __restrict__ src,
                            const float* __restrict__ bias,
                            _Float16* __restrict__ dst,
                            int srcRows, int srcCols, int dstRows, int dstLd) {
  int i = blockIdx.x * blockDim.x + threadIdx.x;
  if (i >= dstRows * dstLd) return;
  int r = i / dstLd, c = i - r * dstLd;
  float v = 0.0f;
  if (r < srcRows && c < srcCols) {
    v = src[(size_t)r * srcCols + c];
    if (bias) v += bias[c];
  }
  dst[i] = (_Float16)v;
}

// ---------------- WMMA GEMM ----------------
// C[M x Ncols] = A[M x K] * B[K x Ncols], A/B f16 row-major (K padded to
// Ksteps*32, B rows >= Ksteps*32, B cols = Ntiles*16 >= Ncols, zero padded).
// One wave computes one 16x16 C tile.
// A frag: lane m=L%16 holds row m; khalf=(L>=16)?8:0; two b128 loads cover
//         K = khalf..khalf+7 and khalf+16..khalf+23.      (ISA 7.12.2)
// B frag: lane L holds B col n=L%16, same K order as A (strided scalar loads).
// C:      VGPR v, lane L -> row v+8*(L>=16), col L%16.
__global__ __launch_bounds__(256) void wmma_gemm(
    const _Float16* __restrict__ A, int lda,
    const _Float16* __restrict__ B, int ldb,
    float* __restrict__ C, int ldc,
    int M, int Ntiles, int Ncols, int Ksteps) {
  int gid  = blockIdx.x * blockDim.x + threadIdx.x;
  int wave = gid >> 5;
  int lane = gid & 31;
  int mtiles = M >> 4;                 // M is a multiple of 16 (N = 50000)
  if (wave >= mtiles * Ntiles) return;
  int mt = wave / Ntiles;
  int nt = wave - mt * Ntiles;
  int row0 = mt << 4, col0 = nt << 4;
  int m     = lane & 15;
  int khalf = (lane >> 4) << 3;        // 0 or 8

  const _Float16* aRow = A + (size_t)(row0 + m) * lda + khalf;
  const _Float16* bRow = B + (size_t)khalf * ldb + col0 + m;

  v8f acc = {};
  for (int ks = 0; ks < Ksteps; ++ks) {
    ABFrag a, b;
    const _Float16* ap = aRow + ks * 32;
    a.f[0] = *(const float4*)(const void*)(ap);
    a.f[1] = *(const float4*)(const void*)(ap + 16);
    const _Float16* bp = bRow + (size_t)(ks * 32) * ldb;
    for (int s = 0; s < 8; ++s) {
      b.v[s]     = bp[(size_t)s * ldb];
      b.v[8 + s] = bp[(size_t)(16 + s) * ldb];
    }
    acc = __builtin_amdgcn_wmma_f32_16x16x32_f16(
        /*neg_a=*/false, a.v, /*neg_b=*/false, b.v,
        /*c_mod=*/(short)0, acc, /*reuse_a=*/false, /*reuse_b=*/false);
  }

  int crow = row0 + khalf;
  int ccol = col0 + m;
  if (ccol < Ncols) {
    #pragma unroll
    for (int v = 0; v < 8; ++v)
      C[(size_t)(crow + v) * ldc + ccol] = acc[v];
  }
}

// ---------------- GAT pieces ----------------

// el[n,h] = sum_f feat[n,h,f]*attn_l[h,f];  er likewise.
__global__ void attn_scores(const float* __restrict__ feat, int ldf,
                            const float* __restrict__ al,
                            const float* __restrict__ ar,
                            float* __restrict__ el, float* __restrict__ er,
                            int n, int H, int F) {
  int i = blockIdx.x * blockDim.x + threadIdx.x;
  if (i >= n * H) return;
  int node = i / H, h = i - node * H;
  const float* fp = feat + (size_t)node * ldf + h * F;
  const float* alp = al + h * F;
  const float* arp = ar + h * F;
  float sl = 0.0f, sr = 0.0f;
  for (int f = 0; f < F; ++f) { float v = fp[f]; sl += v * alp[f]; sr += v * arp[f]; }
  el[i] = sl; er[i] = sr;
}

// monotone float atomic max via int/uint ordering trick
__device__ inline void atomicMaxF(float* addr, float val) {
  if (val >= 0.0f) atomicMax((int*)addr, __float_as_int(val));
  else             atomicMin((unsigned int*)addr, __float_as_uint(val));
}

__global__ void edge_score(const int* __restrict__ src, const int* __restrict__ dst,
                           const float* __restrict__ el, const float* __restrict__ er,
                           float* __restrict__ ebuf, float* __restrict__ m,
                           int E, int H) {
  int i = blockIdx.x * blockDim.x + threadIdx.x;
  if (i >= E * H) return;
  int e = i / H, h = i - e * H;
  float v = el[src[e] * H + h] + er[dst[e] * H + h];
  v = (v > 0.0f) ? v : 0.2f * v;          // leaky_relu(0.2)
  ebuf[i] = v;
  atomicMaxF(&m[dst[e] * H + h], v);
}

__global__ void edge_exp(const int* __restrict__ dst,
                         float* __restrict__ ebuf, const float* __restrict__ m,
                         float* __restrict__ s, int E, int H) {
  int i = blockIdx.x * blockDim.x + threadIdx.x;
  if (i >= E * H) return;
  int e = i / H, h = i - e * H;
  float a = __expf(ebuf[i] - m[dst[e] * H + h]);
  ebuf[i] = a;                             // reuse buffer: e -> a in place
  atomicAdd(&s[dst[e] * H + h], a);
}

// out[dst] += feat[src] * alpha ; thread handles one (edge, 4-feature chunk)
__global__ void edge_aggr(const int* __restrict__ src, const int* __restrict__ dst,
                          const float* __restrict__ a, const float* __restrict__ s,
                          const float* __restrict__ feat, int ldf,
                          float* __restrict__ out, int E, int H, int F) {
  int chunks = (H * F) >> 2;
  int i = blockIdx.x * blockDim.x + threadIdx.x;
  if (i >= E * chunks) return;
  int e = i / chunks;
  int j = (i - e * chunks) << 2;           // position in [0, H*F)
  int h = j / F;
  int d = dst[e], sn = src[e];
  float alpha = a[(size_t)e * H + h] / s[(size_t)d * H + h];
  const float4 fv = *(const float4*)(const void*)(feat + (size_t)sn * ldf + h * F + (j - h * F));
  float* op = out + (size_t)d * (H * F) + j;
  atomicAdd(op + 0, fv.x * alpha);
  atomicAdd(op + 1, fv.y * alpha);
  atomicAdd(op + 2, fv.z * alpha);
  atomicAdd(op + 3, fv.w * alpha);
}

// ---------------- readout ----------------

__global__ void graph_count(const int* __restrict__ gid, float* __restrict__ cnt, int n) {
  int i = blockIdx.x * blockDim.x + threadIdx.x;
  if (i < n) atomicAdd(&cnt[gid[i]], 1.0f);
}

__global__ void graph_accum(const float* __restrict__ h, const float* __restrict__ bias,
                            const int* __restrict__ gid, float* __restrict__ tot,
                            int n, int F, int totLd, int colOff) {
  int i = blockIdx.x * blockDim.x + threadIdx.x;
  if (i >= n * F) return;
  int node = i / F, f = i - node * F;
  atomicAdd(&tot[(size_t)gid[node] * totLd + colOff + f], h[i] + bias[f]);
}

__global__ void graph_norm(float* __restrict__ r, const float* __restrict__ cnt,
                           int G_, int D) {
  int i = blockIdx.x * blockDim.x + threadIdx.x;
  if (i >= G_ * D) return;
  float c = cnt[i / D];
  r[i] = r[i] / (c > 1.0f ? c : 1.0f);
}

__global__ void fc_kernel(const float* __restrict__ r, const float* __restrict__ W,
                          const float* __restrict__ b, float* __restrict__ o,
                          int G_, int Din, int Dout, int doRelu) {
  int i = blockIdx.x * blockDim.x + threadIdx.x;
  if (i >= G_ * Dout) return;
  int g = i / Dout, j = i - g * Dout;
  float acc = b[j];
  const float* rp = r + (size_t)g * Din;
  for (int k = 0; k < Din; ++k) acc += rp[k] * W[(size_t)k * Dout + j];
  o[i] = (doRelu && acc < 0.0f) ? 0.0f : acc;
}

// ---------------- host orchestration ----------------

extern "C" void kernel_launch(void* const* d_in, const int* in_sizes, int n_in,
                              void* d_out, int out_size, void* d_ws, size_t ws_size,
                              hipStream_t stream) {
  (void)in_sizes; (void)n_in; (void)out_size; (void)ws_size;
  const float* xs[3] = {(const float*)d_in[0], (const float*)d_in[1], (const float*)d_in[2]};
  const int*   src  = (const int*)d_in[3];
  const int*   dst  = (const int*)d_in[4];
  const int*   gid  = (const int*)d_in[5];
  const float* W1   = (const float*)d_in[6];
  const float* al1  = (const float*)d_in[7];
  const float* ar1  = (const float*)d_in[8];
  const float* b1   = (const float*)d_in[9];
  const float* W2   = (const float*)d_in[10];
  const float* al2  = (const float*)d_in[11];
  const float* ar2  = (const float*)d_in[12];
  const float* b2   = (const float*)d_in[13];
  const float* Wfc  = (const float*)d_in[14];
  const float* bfc  = (const float*)d_in[15];
  const float* Wcls = (const float*)d_in[16];
  const float* bcls = (const float*)d_in[17];
  float* out = (float*)d_out;

  const int N = 50000, E = 800000, G = 500, F = 100, C = 55;
  const int K1 = 128, NO1 = 208;   // layer1: K 100->128, out cols 200->208 (13 tiles)
  const int K2 = 224, NO2 = 112;   // layer2: K 200->224, out cols 100->112 (7 tiles)

  char* ws = (char*)d_ws;
  size_t off = 0;
  auto alloc = [&](size_t bytes) -> char* {
    char* p = ws + off; off += (bytes + 255) & ~(size_t)255; return p;
  };
  _Float16* W1h  = (_Float16*)alloc((size_t)K1 * NO1 * 2);
  _Float16* W2h  = (_Float16*)alloc((size_t)K2 * NO2 * 2);
  _Float16* x16  = (_Float16*)alloc((size_t)N * K1 * 2);   // layer1 A
  _Float16* h16  = (_Float16*)alloc((size_t)N * K2 * 2);   // layer2 A
  float* feat = (float*)alloc((size_t)N * NO1 * 4);        // feat1 (ld 208) / feat2 (ld 112)
  float* agg  = (float*)alloc((size_t)N * 200 * 4);        // out1 (ld 200) / out2 (ld 100)
  float* ebuf = (float*)alloc((size_t)E * 2 * 4);          // edge e -> a, both layers
  float* el   = (float*)alloc((size_t)N * 2 * 4);
  float* er   = (float*)alloc((size_t)N * 2 * 4);
  float* mv   = (float*)alloc((size_t)N * 2 * 4);
  float* sv   = (float*)alloc((size_t)N * 2 * 4);
  float* tot  = (float*)alloc((size_t)G * 3 * F * 4);
  float* cnt  = (float*)alloc((size_t)G * 4);
  float* fco  = (float*)alloc((size_t)G * F * 4);

  dim3 blk(256);
  auto grid = [](long long n) { return dim3((unsigned)((n + 255) / 256)); };
  const float NEG_INF = -__builtin_inff();

  // pack weights to padded f16 once
  cvt_pad_f16<<<grid((long long)K1 * NO1), blk, 0, stream>>>(W1, nullptr, W1h, 100, 200, K1, NO1);
  cvt_pad_f16<<<grid((long long)K2 * NO2), blk, 0, stream>>>(W2, nullptr, W2h, 200, 100, K2, NO2);

  fill_f32<<<grid(G * 3 * F), blk, 0, stream>>>(tot, 0.0f, G * 3 * F);
  fill_f32<<<grid(G), blk, 0, stream>>>(cnt, 0.0f, G);
  graph_count<<<grid(N), blk, 0, stream>>>(gid, cnt, N);

  for (int b = 0; b < 3; ++b) {
    // ---- GATConv layer 1 (H=2) ----
    cvt_pad_f16<<<grid((long long)N * K1), blk, 0, stream>>>(xs[b], nullptr, x16, N, F, N, K1);
    wmma_gemm<<<grid((long long)(N / 16) * 13 * 32), blk, 0, stream>>>(
        x16, K1, W1h, NO1, feat, NO1, N, 13, 200, K1 / 32);
    attn_scores<<<grid((long long)N * 2), blk, 0, stream>>>(feat, NO1, al1, ar1, el, er, N, 2, F);
    fill_f32<<<grid(N * 2), blk, 0, stream>>>(mv, NEG_INF, N * 2);
    fill_f32<<<grid(N * 2), blk, 0, stream>>>(sv, 0.0f, N * 2);
    fill_f32<<<grid((long long)N * 200), blk, 0, stream>>>(agg, 0.0f, N * 200);
    edge_score<<<grid((long long)E * 2), blk, 0, stream>>>(src, dst, el, er, ebuf, mv, E, 2);
    edge_exp<<<grid((long long)E * 2), blk, 0, stream>>>(dst, ebuf, mv, sv, E, 2);
    edge_aggr<<<grid((long long)E * 50), blk, 0, stream>>>(src, dst, ebuf, sv, feat, NO1, agg, E, 2, F);

    // ---- GATConv layer 2 (H=1); input = agg + b1 ----
    cvt_pad_f16<<<grid((long long)N * K2), blk, 0, stream>>>(agg, b1, h16, N, 200, N, K2);
    wmma_gemm<<<grid((long long)(N / 16) * 7 * 32), blk, 0, stream>>>(
        h16, K2, W2h, NO2, feat, NO2, N, 7, 100, K2 / 32);
    attn_scores<<<grid(N), blk, 0, stream>>>(feat, NO2, al2, ar2, el, er, N, 1, F);
    fill_f32<<<grid(N), blk, 0, stream>>>(mv, NEG_INF, N);
    fill_f32<<<grid(N), blk, 0, stream>>>(sv, 0.0f, N);
    fill_f32<<<grid((long long)N * 100), blk, 0, stream>>>(agg, 0.0f, N * 100);
    edge_score<<<grid(E), blk, 0, stream>>>(src, dst, el, er, ebuf, mv, E, 1);
    edge_exp<<<grid(E), blk, 0, stream>>>(dst, ebuf, mv, sv, E, 1);
    edge_aggr<<<grid((long long)E * 25), blk, 0, stream>>>(src, dst, ebuf, sv, feat, NO2, agg, E, 1, F);

    // ---- graph mean accumulate (adds layer-2 bias) ----
    graph_accum<<<grid((long long)N * F), blk, 0, stream>>>(agg, b2, gid, tot, N, F, 3 * F, b * F);
  }

  // ---- head MLP ----
  graph_norm<<<grid(G * 3 * F), blk, 0, stream>>>(tot, cnt, G, 3 * F);
  fc_kernel<<<grid(G * F), blk, 0, stream>>>(tot, Wfc, bfc, fco, G, 3 * F, F, 1);
  fc_kernel<<<grid(G * C), blk, 0, stream>>>(fco, Wcls, bcls, out, G, F, C, 0);
}
